// MultiSelfAttention_22789096472598
// MI455X (gfx1250) — compile-verified, matrix operand order checked by reading, not run on hardware
//
#include <hip/hip_runtime.h>
#include <hip/hip_bf16.h>

#define BB 2
#define SS 2048
#define EE 1024
#define HH 16
#define DD 64
#define MM (BB*SS)

typedef __attribute__((ext_vector_type(16))) __bf16 v16bf;
typedef __attribute__((ext_vector_type(8)))  __bf16 v8bf;
typedef __attribute__((ext_vector_type(8)))  float  v8f;

static __device__ __forceinline__ __bf16 tobf(float x) { return (__bf16)x; }

// low 32 bits of a generic shared-memory address = LDS byte offset
static __device__ __forceinline__ unsigned lds_addr(const void* p) {
  return (unsigned)(uintptr_t)p;
}

static __device__ __forceinline__ v8bf cvt8(float4 a, float4 b) {
  v8bf r;
  r[0] = tobf(a.x); r[1] = tobf(a.y); r[2] = tobf(a.z); r[3] = tobf(a.w);
  r[4] = tobf(b.x); r[5] = tobf(b.y); r[6] = tobf(b.z); r[7] = tobf(b.w);
  return r;
}

static __device__ __forceinline__ v16bf join8(v8bf lo, v8bf hi) {
  return __builtin_shufflevector(lo, hi, 0,1,2,3,4,5,6,7,8,9,10,11,12,13,14,15);
}

// ---------------------------------------------------------------------------
// One-shot f32 -> bf16 conversion (weights): removes all B-side converts
// from the GEMM hot loops.
// ---------------------------------------------------------------------------
__global__ __launch_bounds__(256) void f32_to_bf16_kernel(
    const float* __restrict__ in, __bf16* __restrict__ out, int n)
{
  const int i = (blockIdx.x * 256 + threadIdx.x) * 8;
  if (i >= n) return;
  *(v8bf*)(out + i) = cvt8(*(const float4*)(in + i), *(const float4*)(in + i + 4));
}

// ---------------------------------------------------------------------------
// QKV projection: out[b,h,s,d] = bf16( X[m,:] . W[n,:] + bias[n] )
// A: f32 activations (converted in-loop, amortized 4x). B: pre-converted bf16.
// Each wave computes a 16(M) x 64(N) strip.
// ---------------------------------------------------------------------------
__global__ __launch_bounds__(256) void qkv_proj_kernel(
    const float*  __restrict__ X,    // [M,K] f32
    const __bf16* __restrict__ W,    // [N,K] bf16  (X @ W^T)
    const float*  __restrict__ bias, // [N]
    __bf16* __restrict__ out)        // [B,H,S,D] bf16
{
  const int lane = threadIdx.x & 31;
  const int wave = threadIdx.x >> 5;
  const int tile = blockIdx.x * 8 + wave;       // 4096 wave-strips
  const int tn4  = tile & 15;                   // N strip: 64 cols
  const int tm   = tile >> 4;                   // M tile: 16 rows

  const int r     = lane & 15;
  const int half8 = (lane >> 4) << 3;

  const float*  Xrow = X + (size_t)(tm * 16 + r) * EE;
  const __bf16* Wr0  = W + (size_t)(tn4 * 64 + r) * EE;

  v8f c[4] = {{}, {}, {}, {}};
  for (int kk = 0; kk < EE; kk += 32) {
    __builtin_prefetch(Xrow + kk + 64, 0, 0);
    const int kb = kk + half8;
    const v16bf a = join8(
        cvt8(*(const float4*)(Xrow + kb),      *(const float4*)(Xrow + kb + 4)),
        cvt8(*(const float4*)(Xrow + kb + 16), *(const float4*)(Xrow + kb + 20)));
#pragma unroll
    for (int t = 0; t < 4; ++t) {
      const __bf16* Wrow = Wr0 + (size_t)(t * 16) * EE;
      const v16bf bfrag = join8(*(const v8bf*)(Wrow + kb),
                                *(const v8bf*)(Wrow + kb + 16));
      c[t] = __builtin_amdgcn_wmma_f32_16x16x32_bf16(false, a, false, bfrag,
                                                     (short)0, c[t], false, false);
    }
  }

  // epilogue: h = tn4 (strip is 64-aligned), d = t*16 + r, no divisions
  const int mbase = tm * 16 + half8;
  const int b     = mbase >> 11;                 // / SS
  const int sbase = mbase & (SS - 1);
  const size_t obase = ((size_t)(b * HH + tn4) * SS + sbase) * DD;
#pragma unroll
  for (int t = 0; t < 4; ++t) {
    const float bv = bias[tn4 * 64 + t * 16 + r];
#pragma unroll
    for (int i = 0; i < 8; ++i)
      out[obase + (size_t)i * DD + t * 16 + r] = tobf(c[t][i] + bv);
  }
}

// ---------------------------------------------------------------------------
// Flash-style attention: per (b,h), 128 query rows/block (16 per wave).
// K/V chunks (32 keys) staged via GLOBAL_LOAD_ASYNC_TO_LDS_B128 (ASYNCcnt).
// ---------------------------------------------------------------------------
__global__ __launch_bounds__(256) void attention_kernel(
    const __bf16* __restrict__ Q,   // [B,H,S,D]
    const __bf16* __restrict__ K,   // [B,H,S,D]
    const __bf16* __restrict__ V,   // [B,H,S,D]
    const int*    __restrict__ mask,// [B,S]
    __bf16* __restrict__ O)         // [B,S,E]
{
  __shared__ __align__(16) __bf16 Kt[32 * DD];    // [32 keys][64 d] row-major
  __shared__ __align__(16) __bf16 Vt[32 * DD];    // [32 keys][64 d] row-major
  __shared__ __align__(16) __bf16 Pl[8][16 * 32]; // per-wave P tile

  const int lane = threadIdx.x & 31;
  const int wave = threadIdx.x >> 5;
  const int b = blockIdx.z;
  const int h = blockIdx.y;
  const int q0 = blockIdx.x * 128 + wave * 16;

  const size_t headoff = (size_t)(b * HH + h) * SS * DD;
  const __bf16* Qh = Q + headoff;
  const __bf16* Kh = K + headoff;
  const __bf16* Vh = V + headoff;
  const int* maskb = mask + b * SS;

  const int r     = lane & 15;
  const int half8 = (lane >> 4) << 3;

  // Q A-fragments (two 32-wide K-chunks over D=64), loaded once
  v16bf qa[2];
  {
    const __bf16* qrow = Qh + (size_t)(q0 + r) * DD;
#pragma unroll
    for (int kc = 0; kc < 2; ++kc) {
      const int kb = kc * 32 + half8;
      qa[kc] = join8(*(const v8bf*)(qrow + kb), *(const v8bf*)(qrow + kb + 16));
    }
  }

  v8f acc[4] = {{}, {}, {}, {}};
  float rowmax[8], rowsum[8];
#pragma unroll
  for (int i = 0; i < 8; ++i) { rowmax[i] = -1e30f; rowsum[i] = 0.0f; }

  const float scale = 0.125f;   // 1/sqrt(64)

  // per-thread async-copy addresses: thread t moves 16B (8 bf16)
  const unsigned ldsK = lds_addr(Kt) + threadIdx.x * 16u;
  const unsigned ldsV = lds_addr(Vt) + threadIdx.x * 16u;

  for (int j0 = 0; j0 < SS; j0 += 32) {
    __syncthreads();
    {
      const __bf16* gk = Kh + (size_t)j0 * DD + threadIdx.x * 8;
      const __bf16* gv = Vh + (size_t)j0 * DD + threadIdx.x * 8;
      asm volatile("global_load_async_to_lds_b128 %0, %1, off"
                   :: "v"(ldsK), "v"(gk) : "memory");
      asm volatile("global_load_async_to_lds_b128 %0, %1, off"
                   :: "v"(ldsV), "v"(gv) : "memory");
      asm volatile("s_wait_asynccnt 0x0" ::: "memory");
    }
    __syncthreads();

    // scores S = Q . K^T for the two 16-key halves
    float sA[8], sB[8];
#pragma unroll
    for (int jj = 0; jj < 2; ++jj) {
      v8f c = {};
#pragma unroll
      for (int kc = 0; kc < 2; ++kc) {
        const int kb = kc * 32 + half8;
        const __bf16* krow = &Kt[(jj * 16 + r) * DD];  // B col = key r
        const v16bf bf = join8(*(const v8bf*)(krow + kb),
                               *(const v8bf*)(krow + kb + 16));
        c = __builtin_amdgcn_wmma_f32_16x16x32_bf16(false, qa[kc], false, bf,
                                                    (short)0, c, false, false);
      }
#pragma unroll
      for (int i = 0; i < 8; ++i) { if (jj == 0) sA[i] = c[i]; else sB[i] = c[i]; }
    }

    // mask, scale, online softmax (reductions stay within 16-lane halves)
    const int mj0 = maskb[j0 + r];
    const int mj1 = maskb[j0 + 16 + r];
#pragma unroll
    for (int i = 0; i < 8; ++i) {
      sA[i] = (mj0 ? sA[i] : -1e20f) * scale;
      sB[i] = (mj1 ? sB[i] : -1e20f) * scale;
      float t = fmaxf(sA[i], sB[i]);
#pragma unroll
      for (int x = 1; x < 16; x <<= 1) t = fmaxf(t, __shfl_xor(t, x, 32));
      const float nm  = fmaxf(rowmax[i], t);
      const float fac = __expf(rowmax[i] - nm);
      rowmax[i] = nm;
      const float p0 = __expf(sA[i] - nm);
      const float p1 = __expf(sB[i] - nm);
      float ps = p0 + p1;
#pragma unroll
      for (int x = 1; x < 16; x <<= 1) ps += __shfl_xor(ps, x, 32);
      rowsum[i] = rowsum[i] * fac + ps;
#pragma unroll
      for (int t4 = 0; t4 < 4; ++t4) acc[t4][i] *= fac;
      Pl[wave][(half8 + i) * 32 + r]      = tobf(p0);
      Pl[wave][(half8 + i) * 32 + 16 + r] = tobf(p1);
    }

    // reload P in A layout (same-wave DScnt dependency handled by compiler)
    v16bf pa;
    {
      const __bf16* prow = &Pl[wave][r * 32];
      pa = join8(*(const v8bf*)(prow + half8), *(const v8bf*)(prow + half8 + 16));
    }
    // acc += P . V  (V row-major in LDS: B element (k, d) = Vt[k*64 + d])
#pragma unroll
    for (int t4 = 0; t4 < 4; ++t4) {
      v16bf vb;
      const int d = t4 * 16 + r;
#pragma unroll
      for (int i = 0; i < 4; ++i) {
        vb[2*i]     = Vt[(half8 + 2*i) * DD + d];
        vb[2*i+1]   = Vt[(half8 + 2*i + 1) * DD + d];
        vb[8+2*i]   = Vt[(half8 + 16 + 2*i) * DD + d];
        vb[8+2*i+1] = Vt[(half8 + 16 + 2*i + 1) * DD + d];
      }
      acc[t4] = __builtin_amdgcn_wmma_f32_16x16x32_bf16(false, pa, false, vb,
                                                        (short)0, acc[t4], false, false);
    }
  }

  // normalize and write [B,S,E] bf16
#pragma unroll
  for (int i = 0; i < 8; ++i) {
    const float inv = 1.0f / rowsum[i];
    const size_t ob = ((size_t)b * SS + (q0 + half8 + i)) * EE + h * DD;
#pragma unroll
    for (int t4 = 0; t4 < 4; ++t4)
      O[ob + t4 * 16 + r] = tobf(acc[t4][i] * inv);
  }
}

// ---------------------------------------------------------------------------
// Output projection: out[m,n] = X[m,:] . Wo[n,:] + bo[n]
// Both A (attention output) and B (pre-converted Wo) are bf16: pure b128 loads.
// ---------------------------------------------------------------------------
__global__ __launch_bounds__(256) void out_proj_kernel(
    const __bf16* __restrict__ X,   // [M,K] bf16
    const __bf16* __restrict__ W,   // [N,K] bf16
    const float*  __restrict__ bias,
    float* __restrict__ out)        // [M,N] f32
{
  const int lane = threadIdx.x & 31;
  const int wave = threadIdx.x >> 5;
  const int tile = blockIdx.x * 8 + wave;
  const int tn4  = tile & 15;
  const int tm   = tile >> 4;

  const int r     = lane & 15;
  const int half8 = (lane >> 4) << 3;

  const __bf16* Xrow = X + (size_t)(tm * 16 + r) * EE;
  const __bf16* Wr0  = W + (size_t)(tn4 * 64 + r) * EE;

  v8f c[4] = {{}, {}, {}, {}};
  for (int kk = 0; kk < EE; kk += 32) {
    __builtin_prefetch(Xrow + kk + 64, 0, 0);
    const int kb = kk + half8;
    const v16bf a = join8(*(const v8bf*)(Xrow + kb), *(const v8bf*)(Xrow + kb + 16));
#pragma unroll
    for (int t = 0; t < 4; ++t) {
      const __bf16* Wrow = Wr0 + (size_t)(t * 16) * EE;
      const v16bf bfrag = join8(*(const v8bf*)(Wrow + kb),
                                *(const v8bf*)(Wrow + kb + 16));
      c[t] = __builtin_amdgcn_wmma_f32_16x16x32_bf16(false, a, false, bfrag,
                                                     (short)0, c[t], false, false);
    }
  }

  const int mbase = tm * 16 + half8;
#pragma unroll
  for (int t = 0; t < 4; ++t) {
    const int n = tn4 * 64 + t * 16 + r;
    const float bv = bias[n];
#pragma unroll
    for (int i = 0; i < 8; ++i)
      out[(size_t)(mbase + i) * EE + n] = c[t][i] + bv;
  }
}

// ---------------------------------------------------------------------------
extern "C" void kernel_launch(void* const* d_in, const int* in_sizes, int n_in,
                              void* d_out, int out_size, void* d_ws, size_t ws_size,
                              hipStream_t stream)
{
  const float* queries = (const float*)d_in[0];
  const float* keys    = (const float*)d_in[1];
  // d_in[2] (values) intentionally unused: reference applies Wv to `keys`.
  const int*   mask    = (const int*)d_in[3];
  const float* Wq = (const float*)d_in[4];
  const float* bq = (const float*)d_in[5];
  const float* Wk = (const float*)d_in[6];
  const float* bk = (const float*)d_in[7];
  const float* Wv = (const float*)d_in[8];
  const float* bv = (const float*)d_in[9];
  const float* Wo = (const float*)d_in[10];
  const float* bo = (const float*)d_in[11];

  const size_t elems  = (size_t)MM * EE;  // 4M elems (activations)
  const size_t welems = (size_t)EE * EE;  // 1M elems (weights)
  __bf16* qw  = (__bf16*)d_ws;
  __bf16* kw  = qw  + elems;
  __bf16* vw  = kw  + elems;
  __bf16* aw  = vw  + elems;
  __bf16* wqb = aw  + elems;
  __bf16* wkb = wqb + welems;
  __bf16* wvb = wkb + welems;
  __bf16* wob = wvb + welems;

  dim3 blk(256);

  // one-shot weight conversion to bf16
  dim3 gcvt((int)(welems / (8 * 256)));
  f32_to_bf16_kernel<<<gcvt, blk, 0, stream>>>(Wq, wqb, (int)welems);
  f32_to_bf16_kernel<<<gcvt, blk, 0, stream>>>(Wk, wkb, (int)welems);
  f32_to_bf16_kernel<<<gcvt, blk, 0, stream>>>(Wv, wvb, (int)welems);
  f32_to_bf16_kernel<<<gcvt, blk, 0, stream>>>(Wo, wob, (int)welems);

  // (M/16) * (N/64) wave-strips, 8 waves per block
  dim3 gproj(((MM / 16) * (EE / 64)) / 8);
  qkv_proj_kernel<<<gproj, blk, 0, stream>>>(queries, wqb, bq, qw);
  qkv_proj_kernel<<<gproj, blk, 0, stream>>>(keys,    wkb, bk, kw);
  qkv_proj_kernel<<<gproj, blk, 0, stream>>>(keys,    wvb, bv, vw);

  attention_kernel<<<dim3(SS / 128, HH, BB), blk, 0, stream>>>(qw, kw, vw, mask, aw);

  out_proj_kernel<<<gproj, blk, 0, stream>>>(aw, wob, bo, (float*)d_out);
}
